// RevEncoder_52621939310786
// MI455X (gfx1250) — matrix operand order repaired and checked by reading, not compiled, not run on hardware
//
#include <hip/hip_runtime.h>
#include <hip/hip_bf16.h>

typedef __attribute__((ext_vector_type(16))) _Float16 v16h;
typedef __attribute__((ext_vector_type(8)))  _Float16 v8h;
typedef __attribute__((ext_vector_type(8)))  float    v8f;

#define NWG 16
#define Tn  512
#define Bn  64
#define Hn  512
#define Dn  256
#define Kn  768
#define OUT_SAVED (2*Bn*Hn)   // 65536

// ---------------- ws layout (bytes) ----------------
constexpr size_t OFF_BAR = 0;                               // 256 B barrier counter
constexpr size_t OFF_H   = 256;                             // int32 [64][512] state
constexpr size_t OFF_ZQ  = OFF_H  + (size_t)Bn*Hn*4;        // int32 [64][256] quant gates
constexpr size_t OFF_HF  = OFF_ZQ + (size_t)Bn*Dn*4;        // f16   [64][256] h-half as float
constexpr size_t OFF_RH  = OFF_HF + (size_t)Bn*Dn*2;        // f16   [64][256] r*hf
constexpr size_t OFF_XB  = OFF_RH + (size_t)Bn*Dn*2;        // f16   [2][64][512] x double-buffer
constexpr size_t OFF_WZ1 = OFF_XB + (size_t)2*Bn*Hn*2;      // f16   [L][512][768] Wz1^T
constexpr size_t OFF_WG1 = OFF_WZ1 + (size_t)2*Hn*Kn*2;     // f16   [L][256][768] Wg1^T
constexpr size_t OFF_WZ2 = OFF_WG1 + (size_t)2*Dn*Kn*2;     // f16   [L][512][768] Wz2^T
constexpr size_t OFF_WG2 = OFF_WZ2 + (size_t)2*Hn*Kn*2;     // f16   [L][256][768] Wg2^T
constexpr size_t OFF_XS  = OFF_WG2 + (size_t)2*Dn*Kn*2;     // f16   [512][64][512] layer-0 h

__device__ __forceinline__ v8h ld8h(const _Float16* p) { return *(const v8h*)p; }
__device__ __forceinline__ v16h hcat(v8h a, v8h b) {
  return __builtin_shufflevector(a, b, 0,1,2,3,4,5,6,7,8,9,10,11,12,13,14,15);
}

// hardware-transcendental activations (v_exp_f32 + v_rcp_f32, no IEEE divide expansion)
__device__ __forceinline__ float fast_sigmoid(float x) {
  return __builtin_amdgcn_rcpf(1.0f + __expf(-x));
}
__device__ __forceinline__ float fast_tanh(float x) {   // 1 - 2/(e^{2x}+1); exact limits at +/-inf
  return 1.0f - 2.0f * __builtin_amdgcn_rcpf(1.0f + __expf(2.0f * x));
}

// 16x16 output tile over K-step range [kt0,kt1), K-step = 32. K = 768 = 512 x-part + 256 h-part.
// A layout (16-bit 16x32): lane=(lo,hi): halves k = kb+hi*8..+7 and kb+16+hi*8..+7
// B layout (16-bit 32x16): lane=(lo,hi): halves k = kb+hi*8..+7 and kb+16+hi*8..+7, col n = lo
__device__ __forceinline__ v8f gemm_tile_range(const _Float16* __restrict__ X,
                                               const _Float16* __restrict__ HFp,
                                               const _Float16* __restrict__ wrow,
                                               int m0, int lo, int hi, int kt0, int kt1)
{
  const _Float16* ar = X   + (size_t)(m0 + lo) * Hn + hi*8;        // x rows (kt 0..15)
  const _Float16* hr = HFp + (size_t)(m0 + lo) * Dn + hi*8 - 512;  // h rows, shifted: +kt*32 (kt 16..23)
  const _Float16* br = wrow + hi*8;
  v8f acc = {};
  #pragma unroll 4
  for (int kt = kt0; kt < kt1; ++kt) {
    const _Float16* arow = ((kt < 16) ? ar : hr) + kt*32;
    v16h a = hcat(ld8h(arow), ld8h(arow + 16));
    const _Float16* brow = br + kt*32;
    v16h b = hcat(ld8h(brow), ld8h(brow + 16));
    acc = __builtin_amdgcn_wmma_f32_16x16x32_f16(false, a, false, b, (short)0, acc, false, false);
  }
  return acc;
}

// sigmoid gate epilogue: cols<256 -> quantized forget gate, cols>=256 -> r*hf
__device__ __forceinline__ void epil_z(v8f acc, int n0, int m0, int lo, int hi,
    const float* __restrict__ bz, int* __restrict__ zq,
    const _Float16* __restrict__ HF, _Float16* __restrict__ RH)
{
  int col = n0 + lo;
  float bias = bz[col];
  #pragma unroll
  for (int r = 0; r < 8; ++r) {
    int m = m0 + r + hi*8;               // C/D layout: VGPR r -> M=r (lanes 0-15), r+8 (16-31)
    float s = fast_sigmoid(acc[r] + bias);
    if (col < Dn) {
      float zzv = 0.875f * s + 0.125f;   // MAX_FORGET*z + (1-MAX_FORGET)
      int q = (int)floorf(zzv * 1024.0f);
      q = q < 1 ? 1 : (q > 1024 ? 1024 : q);
      zq[m*Dn + col] = q;
    } else {
      int c2 = col - Dn;
      RH[m*Dn + c2] = (_Float16)(s * (float)HF[m*Dn + c2]);
    }
  }
}

// tanh gate epilogue + fixed-point reversible state update
__device__ __forceinline__ void epil_g(v8f acc, int n0, int m0, int lo, int hi,
    const float* __restrict__ bg, const int* __restrict__ zq,
    int* __restrict__ hstate, _Float16* __restrict__ HF,
    _Float16* __restrict__ xst, int* __restrict__ savedp,
    const int* __restrict__ lens, int t, int hofs)
{
  int col = n0 + lo;
  float bias = bg[col];
  #pragma unroll
  for (int r = 0; r < 8; ++r) {
    int m = m0 + r + hi*8;
    float g = fast_tanh(acc[r] + bias);
    int sc  = hofs + col;
    int idx = m*Hn + sc;
    int hv  = hstate[idx];
    if (t < lens[m]) {
      int q   = zq[m*Dn + col];
      int fm  = (hv >> 10) * q + (((hv & 1023) * q) >> 10);   // _fixed_mul
      float zf = (float)q * (1.0f / 1024.0f);
      int add = __float2int_rn((1.0f - zf) * g * 8388608.0f); // round((1-z)*g*2^23)
      hv = fm + add;
      hstate[idx] = hv;
    }
    _Float16 hf = (_Float16)((float)hv * (1.0f / 8388608.0f));
    HF[m*Dn + col] = hf;
    if (xst)    xst[((size_t)t*Bn + m)*Hn + sc] = hf;
    if (savedp && col < 100) savedp[((t+1)*Bn + m)*100 + col] = hv;
  }
}

// ------------- prep: weight transpose->f16, x(t=0) gather, zero bar/saved[0] -------------
__global__ void rev_prep(
    const float* __restrict__ Wz1, const float* __restrict__ Wg1,
    const float* __restrict__ Wz2, const float* __restrict__ Wg2,
    const int* __restrict__ seq, const float* __restrict__ emb,
    _Float16* __restrict__ wz1t, _Float16* __restrict__ wg1t,
    _Float16* __restrict__ wz2t, _Float16* __restrict__ wg2t,
    _Float16* __restrict__ xbuf, int* __restrict__ out,
    unsigned* __restrict__ bar)
{
  long i0 = (long)blockIdx.x * blockDim.x + threadIdx.x;
  long stride = (long)gridDim.x * blockDim.x;
  const long NZ = 2L * Hn * Kn;
  const long NG = 2L * Dn * Kn;
  for (long i = i0; i < NZ; i += stride) {              // [l][n][k] <- [l][k][n]
    long l = i / (Hn*Kn), rr = i % (Hn*Kn), n = rr / Kn, k = rr % Kn;
    wz1t[i] = (_Float16)Wz1[(l*Kn + k)*Hn + n];
    wz2t[i] = (_Float16)Wz2[(l*Kn + k)*Hn + n];
  }
  for (long i = i0; i < NG; i += stride) {
    long l = i / (Dn*Kn), rr = i % (Dn*Kn), n = rr / Kn, k = rr % Kn;
    wg1t[i] = (_Float16)Wg1[(l*Kn + k)*Dn + n];
    wg2t[i] = (_Float16)Wg2[(l*Kn + k)*Dn + n];
  }
  for (long i = i0; i < (long)Bn*Hn; i += stride) {     // x for t=0 (layer 0)
    int b = (int)(i >> 9), c = (int)(i & 511);
    xbuf[i] = (_Float16)emb[(long)seq[b]*Hn + c];
  }
  for (long i = i0; i < (long)Bn*100; i += stride)      // saved[0] = h0 = 0
    out[OUT_SAVED + i] = 0;
  if (i0 < 64) bar[i0] = 0;                             // reset device-wide barrier
}

// ------------- main persistent cooperative kernel -------------
__global__ __launch_bounds__(256, 1) void rev_main(
    const int* __restrict__ seq, const int* __restrict__ lens,
    const float* __restrict__ emb,
    const float* __restrict__ bz1, const float* __restrict__ bg1,
    const float* __restrict__ bz2, const float* __restrict__ bg2,
    const _Float16* __restrict__ wz1t, const _Float16* __restrict__ wg1t,
    const _Float16* __restrict__ wz2t, const _Float16* __restrict__ wg2t,
    int* __restrict__ hstate, int* __restrict__ zq,
    _Float16* __restrict__ HF, _Float16* __restrict__ RH,
    _Float16* __restrict__ xbuf, _Float16* __restrict__ xstore,
    unsigned* __restrict__ bar, int* __restrict__ out)
{
  __shared__ float smem[4][256];   // per-m-tile 16x16 f32 partials for G-phase K-split reduce

  const int w    = blockIdx.x;
  const int tid  = threadIdx.x;
  const int lane = tid & 31, wave = tid >> 5;
  const int lo   = lane & 15, hi = lane >> 4;
  unsigned bcnt = 0;

  auto gsync = [&]() {
    __syncthreads();
    ++bcnt;
    if (tid == 0) {
      __threadfence();
      atomicAdd(bar, 1u);
      while (atomicAdd(bar, 0u) < bcnt * NWG) __builtin_amdgcn_s_sleep(4);
    }
    __syncthreads();
    __threadfence();
  };

  // G-phase body: all 8 waves K-split (waves 0-3: K 0..383, waves 4-7: K 384..767),
  // LDS reduce, waves 0-3 run epilogue while waves 4-7 gather next x (layer 0).
  auto g_phase = [&](const _Float16* X, const _Float16* WG, const float* BG,
                     int t, int l, int hofs, bool do_saved) {
    int mt = wave & 3, half = wave >> 2;
    int n0 = 16*w, m0 = 16*mt;
    v8f acc = gemm_tile_range(X, RH, WG + (size_t)(n0 + lo)*Kn, m0, lo, hi,
                              half ? 12 : 0, half ? 24 : 12);
    if (half) {
      #pragma unroll
      for (int r = 0; r < 8; ++r) smem[mt][lane*8 + r] = acc[r];
    }
    __syncthreads();
    if (!half) {
      #pragma unroll
      for (int r = 0; r < 8; ++r) acc[r] += smem[mt][lane*8 + r];
      epil_g(acc, n0, m0, lo, hi, BG, zq, hstate, HF,
             (l == 0) ? xstore : (_Float16*)nullptr,
             do_saved ? (out + OUT_SAVED) : (int*)nullptr, lens, t, hofs);
    } else if (hofs == 0 && l == 0 && t + 1 < Tn) {   // gather next x during G1 only
      int b = 4*w + (wave - 4);
      int tok = seq[(t+1)*Bn + b];
      const float* er = emb + (size_t)tok*Hn;
      _Float16* xr = xbuf + (size_t)((t+1) & 1)*Bn*Hn + (size_t)b*Hn;
      for (int c = lane; c < Hn; c += 32) xr[c] = (_Float16)er[c];
    }
  };

  for (int l = 0; l < 2; ++l) {
    const float* BZ1 = bz1 + l*Hn;  const float* BG1 = bg1 + l*Dn;
    const float* BZ2 = bz2 + l*Hn;  const float* BG2 = bg2 + l*Dn;
    const _Float16* WZ1 = wz1t + (size_t)l*Hn*Kn;
    const _Float16* WG1 = wg1t + (size_t)l*Dn*Kn;
    const _Float16* WZ2 = wz2t + (size_t)l*Hn*Kn;
    const _Float16* WG2 = wg2t + (size_t)l*Dn*Kn;

    // reset h = 0 and HF (= h2f of h0) = 0 for this layer
    for (int i = w*256 + tid; i < Bn*Hn; i += NWG*256) hstate[i] = 0;
    for (int i = w*256 + tid; i < Bn*Dn; i += NWG*256) HF[i] = (_Float16)0.0f;
    gsync();

    for (int t = 0; t < Tn; ++t) {
      const _Float16* X = (l == 0) ? (xbuf + (size_t)(t & 1)*Bn*Hn)
                                   : (xstore + (size_t)t*Bn*Hn);
      // ---- phase Z1: zr1 = sigmoid([x, h2f] @ Wz1 + bz1), 64x512, 32 cols/WG, 8 tiles/WG
      {
        int mt = wave & 3, nt = wave >> 2;
        int n0 = 32*w + 16*nt, m0 = 16*mt;
        v8f acc = gemm_tile_range(X, HF, WZ1 + (size_t)(n0 + lo)*Kn, m0, lo, hi, 0, 24);
        epil_z(acc, n0, m0, lo, hi, BZ1, zq, HF, RH);
      }
      gsync();
      // ---- phase G1: g1 = tanh([x, r1*h2f] @ Wg1 + bg1); update h1 (K-split + LDS reduce)
      g_phase(X, WG1, BG1, t, l, 0, /*saved=*/(l == 1));
      gsync();
      // ---- phase Z2: zr2 from [x, h1f]
      {
        int mt = wave & 3, nt = wave >> 2;
        int n0 = 32*w + 16*nt, m0 = 16*mt;
        v8f acc = gemm_tile_range(X, HF, WZ2 + (size_t)(n0 + lo)*Kn, m0, lo, hi, 0, 24);
        epil_z(acc, n0, m0, lo, hi, BZ2, zq, HF, RH);
      }
      gsync();
      // ---- phase G2: g2 from [x, r2*h1f]; update h2; HF <- h2f for next step
      g_phase(X, WG2, BG2, t, l, Dn, /*saved=*/false);
      gsync();
    }

    // last_hiddens[l] = h_T
    for (int i = tid; i < Bn*32; i += 256) {
      int b = i >> 5, c = 32*w + (i & 31);
      out[l*Bn*Hn + b*Hn + c] = hstate[b*Hn + c];
    }
    gsync();
  }
}

extern "C" void kernel_launch(void* const* d_in, const int* in_sizes, int n_in,
                              void* d_out, int out_size, void* d_ws, size_t ws_size,
                              hipStream_t stream) {
  const int*   seq  = (const int*)  d_in[0];
  const int*   lens = (const int*)  d_in[1];
  const float* emb  = (const float*)d_in[2];
  const float* Wz1  = (const float*)d_in[3];
  const float* bz1  = (const float*)d_in[4];
  const float* Wg1  = (const float*)d_in[5];
  const float* bg1  = (const float*)d_in[6];
  const float* Wz2  = (const float*)d_in[7];
  const float* bz2  = (const float*)d_in[8];
  const float* Wg2  = (const float*)d_in[9];
  const float* bg2  = (const float*)d_in[10];

  char* ws = (char*)d_ws;
  unsigned* bar    = (unsigned*)(ws + OFF_BAR);
  int*      hstate = (int*)     (ws + OFF_H);
  int*      zq     = (int*)     (ws + OFF_ZQ);
  _Float16* HF     = (_Float16*)(ws + OFF_HF);
  _Float16* RH     = (_Float16*)(ws + OFF_RH);
  _Float16* xbuf   = (_Float16*)(ws + OFF_XB);
  _Float16* wz1t   = (_Float16*)(ws + OFF_WZ1);
  _Float16* wg1t   = (_Float16*)(ws + OFF_WG1);
  _Float16* wz2t   = (_Float16*)(ws + OFF_WZ2);
  _Float16* wg2t   = (_Float16*)(ws + OFF_WG2);
  _Float16* xstore = (_Float16*)(ws + OFF_XS);
  int*      out    = (int*)d_out;

  rev_prep<<<2048, 256, 0, stream>>>(Wz1, Wg1, Wz2, Wg2, seq, emb,
                                     wz1t, wg1t, wz2t, wg2t, xbuf, out, bar);
  rev_main<<<NWG, 256, 0, stream>>>(seq, lens, emb, bz1, bg1, bz2, bg2,
                                    wz1t, wg1t, wz2t, wg2t,
                                    hstate, zq, HF, RH, xbuf, xstore, bar, out);
}
